// deepspeech_LocalDotAttenLayers_38654705664179
// MI455X (gfx1250) — compile-verified, matrix operand order checked by reading, not run on hardware
//
#include <hip/hip_runtime.h>
#include <stdint.h>

// Problem constants (fixed by the reference)
constexpr int kB  = 32;    // batch
constexpr int kT  = 512;   // time steps
constexpr int kIN = 2048;  // input dim
constexpr int kD  = 512;   // embed dim
constexpr int kO  = 29;    // output dim
constexpr int KC  = 64;    // GEMM K chunk

typedef __attribute__((ext_vector_type(16))) _Float16 v16h;
typedef __attribute__((ext_vector_type(8)))  float    v8f;
typedef __attribute__((ext_vector_type(4)))  int      v4i;

union Frag16 { v16h v; uint32_t u[8]; };

// ---- gfx1250 async global->LDS path (graceful fallback) -------------------
#if defined(__has_builtin)
#  if __has_builtin(__builtin_amdgcn_global_load_async_to_lds_b128)
#    define HAVE_ASYNC_COPY 1
#  endif
#endif
#ifndef HAVE_ASYNC_COPY
#  define HAVE_ASYNC_COPY 0
#endif

#if defined(__has_builtin)
#  if __has_builtin(__builtin_amdgcn_s_wait_asynccnt)
#    define WAIT_ASYNC() __builtin_amdgcn_s_wait_asynccnt(0)
#  else
#    define WAIT_ASYNC() asm volatile("s_wait_asynccnt 0x0" ::: "memory")
#  endif
#else
#  define WAIT_ASYNC() asm volatile("s_wait_asynccnt 0x0" ::: "memory")
#endif

// ---------------------------------------------------------------------------
// Kernel 0: W_in [k][n] fp32 -> Wt [n][k] f16 (fragment-friendly transpose).
// Only 4 MB read / 2 MB write: negligible vs the GEMM.
// ---------------------------------------------------------------------------
__global__ __launch_bounds__(256)
void convert_w(const float* __restrict__ W_in, _Float16* __restrict__ Wt) {
  const int base = (blockIdx.x * 256 + threadIdx.x) * 4;
  if (base < kIN * kD) {
    const int k = base / kD;
    const int n = base % kD;
    const float4 f = *(const float4*)(W_in + base);
    Wt[(size_t)(n + 0) * kIN + k] = (_Float16)f.x;
    Wt[(size_t)(n + 1) * kIN + k] = (_Float16)f.y;
    Wt[(size_t)(n + 2) * kIN + k] = (_Float16)f.z;
    Wt[(size_t)(n + 3) * kIN + k] = (_Float16)f.w;
  }
}

// ---------------------------------------------------------------------------
// Kernel A: h = x @ W_in + b_in   (M=16384, K=2048, N=512), f16 WMMA, f32 acc
// Block = 256 threads = 8 waves; block tile 128(M) x 128(N); K chunk = 64.
// Wave w owns rows [w*16, w*16+16) and all 8 N-subtiles; 16 WMMA per stage.
// B tile staged via GLOBAL_LOAD_ASYNC_TO_LDS_B128 (ASYNCcnt-tracked).
// B fragments double-buffered so LDS latency hides under WMMA issue.
// ---------------------------------------------------------------------------
__global__ __launch_bounds__(256)
void in_proj_wmma(const float* __restrict__ x,
                  const _Float16* __restrict__ Wt,   // [n][k] f16
                  const float* __restrict__ b_in,
                  float* __restrict__ h) {
  __shared__ _Float16 As[128 * KC];   // [m][k], 16 KB
  __shared__ _Float16 Bs[128 * KC];   // [n][k], 16 KB

  const int tid  = threadIdx.x;
  const int wave = tid >> 5;          // wave32
  const int lane = tid & 31;
  const int half = lane >> 4;
  const int nl   = lane & 15;
  const int gm0  = blockIdx.y * 128;
  const int gn0  = blockIdx.x * 128;

  // K offset of each packed-pair VGPR in the 16-bit 16x32 fragment layout.
  int kbase[8];
#pragma unroll
  for (int v = 0; v < 8; ++v)
    kbase[v] = ((v < 4) ? (2 * v) : (16 + 2 * (v - 4))) + 8 * half;

  v8f cacc[8];
#pragma unroll
  for (int nt = 0; nt < 8; ++nt) {
    v8f z = {0.f, 0.f, 0.f, 0.f, 0.f, 0.f, 0.f, 0.f};
    cacc[nt] = z;
  }

  for (int kc = 0; kc < kIN; kc += KC) {
    // Stage B tile (128 x 64 f16): async 16-byte global->LDS copies.
#pragma unroll
    for (int q = tid; q < 128 * 8; q += 256) {
      const int n = q >> 3;
      const int k = (q & 7) << 3;
      _Float16* g = const_cast<_Float16*>(Wt + (size_t)(gn0 + n) * kIN + kc + k);
      _Float16* l = &Bs[n * KC + k];
#if HAVE_ASYNC_COPY
      __builtin_amdgcn_global_load_async_to_lds_b128(
          (__attribute__((address_space(1))) v4i*)g,
          (__attribute__((address_space(3))) v4i*)l, 0, 0);
#else
      *(uint4*)l = *(const uint4*)g;
#endif
    }
    // Stage A tile (128 x 64 fp32 -> f16): float4 load, packed uint2 store.
#pragma unroll
    for (int q = tid; q < 128 * 16; q += 256) {
      const int m = q >> 4;
      const int k = (q & 15) << 2;
      const float4 f = *(const float4*)(x + (size_t)(gm0 + m) * kIN + kc + k);
      union { _Float16 hh[2]; uint32_t u; } p0, p1;
      p0.hh[0] = (_Float16)f.x; p0.hh[1] = (_Float16)f.y;
      p1.hh[0] = (_Float16)f.z; p1.hh[1] = (_Float16)f.w;
      uint2 pk; pk.x = p0.u; pk.y = p1.u;
      *(uint2*)&As[m * KC + k] = pk;
    }
    // Prefetch next K-chunk of x (global_prefetch_b8).
    if (kc + KC < kIN && tid < 128)
      __builtin_prefetch(x + (size_t)(gm0 + tid) * kIN + kc + KC, 0, 1);
#if HAVE_ASYNC_COPY
    WAIT_ASYNC();
#endif
    __syncthreads();

#pragma unroll
    for (int s = 0; s < 2; ++s) {
      const int koff = 32 * s;
      Frag16 a;
      const int abase = (wave * 16 + nl) * KC + koff;
#pragma unroll
      for (int v = 0; v < 8; ++v)
        a.u[v] = *(const uint32_t*)&As[abase + kbase[v]];

      // Rotating B-fragment buffers: issue loads for nt+1 before wmma(nt).
      Frag16 bf[2];
      {
        const int bbase = (0 * 16 + nl) * KC + koff;
#pragma unroll
        for (int v = 0; v < 8; ++v)
          bf[0].u[v] = *(const uint32_t*)&Bs[bbase + kbase[v]];
      }
#pragma unroll
      for (int nt = 0; nt < 8; ++nt) {
        if (nt < 7) {
          const int bbase = ((nt + 1) * 16 + nl) * KC + koff;
#pragma unroll
          for (int v = 0; v < 8; ++v)
            bf[(nt + 1) & 1].u[v] = *(const uint32_t*)&Bs[bbase + kbase[v]];
        }
        cacc[nt] = __builtin_amdgcn_wmma_f32_16x16x32_f16(
            false, a.v, false, bf[nt & 1].v, (short)0, cacc[nt], false, false);
      }
    }
    __syncthreads();
  }

  // C/D layout: lane -> N = lane&15; VGPR r -> M = r + 8*(lane>>4).
#pragma unroll
  for (int nt = 0; nt < 8; ++nt) {
    const int col  = gn0 + nt * 16 + nl;
    const float bv = b_in[col];
#pragma unroll
    for (int r = 0; r < 8; ++r) {
      const int row = gm0 + wave * 16 + r + 8 * half;
      h[(size_t)row * kD + col] = cacc[nt][r] + bv;
    }
  }
}

// ---------------------------------------------------------------------------
// Kernel B: sequential local-attention scan. One block per batch element.
// h stays hot in L2 (33.5 MB < 192 MB); windowed re-reads are cheap. The
// scan is latency-bound, so loads are widened to b128/b64.
// ---------------------------------------------------------------------------
__global__ __launch_bounds__(256)
void atten_scan(const float* __restrict__ h,
                const int*   __restrict__ lens,
                const float* __restrict__ W_q,
                const float* __restrict__ b_q,
                const float* __restrict__ W_out,
                const float* __restrict__ b_out,
                const int*   __restrict__ wsz,
                float* __restrict__ outO,   // [B,T,29]
                float* __restrict__ outW) { // [B,T,T]
  __shared__ float q_sh[kD];
  __shared__ float out_sh[kD];
  __shared__ float sc_sh[160];
  __shared__ float w_sh[160];
  __shared__ float query_sh[32];
  __shared__ float red_sh[8];
  __shared__ float stat_sh[2];

  const int tid  = threadIdx.x;
  const int wid  = tid >> 5;
  const int lane = tid & 31;
  const int b    = blockIdx.x;
  const int len  = lens[b];
  const int WSZ  = wsz[0];
  const float scale = 0.044194173824159216f;  // 1/sqrt(512)

  if (tid < kO) query_sh[tid] = (tid == kO - 1) ? 9.0f : 1.0f;
  __syncthreads();

  for (int t = 0; t < kT; ++t) {
    const bool alive = (t < len);
    int lo = t - WSZ; if (lo < 0) lo = 0;
    int hi = t + WSZ;
    const int lim = ((len < kT) ? len : kT) - 1;
    if (hi > lim) hi = lim;
    const int wlen = alive ? (hi - lo + 1) : 0;

    // q = query @ W_q + b_q
    if (alive) {
      for (int d = tid; d < kD; d += 256) {
        float acc = b_q[d];
#pragma unroll
        for (int o = 0; o < kO; ++o)
          acc += query_sh[o] * W_q[o * kD + d];
        q_sh[d] = acc;
      }
    }
    __syncthreads();

    // windowed scores: one wave per j, lanes own 4-wide chunks, b128 loads
    if (alive) {
      for (int j = lo + wid; j <= hi; j += 8) {
        const float* hrow = h + ((size_t)b * kT + j) * kD;
        float acc = 0.f;
#pragma unroll
        for (int c = 0; c < 4; ++c) {
          const int d = c * 128 + lane * 4;
          const float4 hv = *(const float4*)(hrow + d);
          const float4 qv = *(const float4*)(q_sh + d);
          acc += hv.x * qv.x + hv.y * qv.y + hv.z * qv.z + hv.w * qv.w;
        }
#pragma unroll
        for (int off = 16; off; off >>= 1)
          acc += __shfl_xor(acc, off, 32);
        if (lane == 0) sc_sh[j - lo] = acc * scale;
      }
    }
    __syncthreads();

    // block softmax over wlen (<= 129) entries
    float v = (tid < wlen) ? sc_sh[tid] : -3.402823466e38f;
#pragma unroll
    for (int off = 16; off; off >>= 1)
      v = fmaxf(v, __shfl_xor(v, off, 32));
    if (lane == 0) red_sh[wid] = v;
    __syncthreads();
    if (tid == 0) {
      float m = red_sh[0];
#pragma unroll
      for (int i = 1; i < 8; ++i) m = fmaxf(m, red_sh[i]);
      stat_sh[0] = m;
    }
    __syncthreads();
    const float smax = stat_sh[0];
    float e = 0.f;
    if (tid < wlen) { e = __expf(sc_sh[tid] - smax); w_sh[tid] = e; }
    float s = e;
#pragma unroll
    for (int off = 16; off; off >>= 1)
      s += __shfl_xor(s, off, 32);
    if (lane == 0) red_sh[wid] = s;
    __syncthreads();
    if (tid == 0) {
      float sm = 0.f;
#pragma unroll
      for (int i = 0; i < 8; ++i) sm += red_sh[i];
      stat_sh[1] = sm;
    }
    __syncthreads();
    const float inv = (wlen > 0) ? 1.0f / stat_sh[1] : 0.0f;

    // out[d] = sum_j w[j] * h[b, lo+j, d]; thread owns a float2 of d
    {
      const int d = tid << 1;
      float ax = 0.f, ay = 0.f;
      for (int i = 0; i < wlen; ++i) {
        const float2 hv =
            *(const float2*)(h + ((size_t)b * kT + lo + i) * kD + d);
        const float wv = w_sh[i];
        ax += wv * hv.x;
        ay += wv * hv.y;
      }
      out_sh[d]     = ax * inv;
      out_sh[d + 1] = ay * inv;
    }

    // full weights row (zeros outside window / dead steps), float2 stores
    {
      float* wrow = outW + ((size_t)b * kT + t) * kT;
      const int j = tid << 1;
      float2 val; val.x = 0.f; val.y = 0.f;
      if (alive) {
        if (j     >= lo && j     <= hi) val.x = w_sh[j - lo] * inv;
        if (j + 1 >= lo && j + 1 <= hi) val.y = w_sh[j + 1 - lo] * inv;
      }
      *(float2*)(wrow + j) = val;
    }
    __syncthreads();

    // new_query = out @ W_out + b_out ; also the per-step output row
    for (int o = wid; o < kO; o += 8) {
      float acc = 0.f;
      for (int d = lane; d < kD; d += 32)
        acc += out_sh[d] * W_out[d * kO + o];
#pragma unroll
      for (int off = 16; off; off >>= 1)
        acc += __shfl_xor(acc, off, 32);
      if (lane == 0) {
        const float nq = acc + b_out[o];
        outO[((size_t)b * kT + t) * kO + o] = nq;
        query_sh[o] = nq;
      }
    }
    __syncthreads();
  }
}

// ---------------------------------------------------------------------------
extern "C" void kernel_launch(void* const* d_in, const int* in_sizes, int n_in,
                              void* d_out, int out_size, void* d_ws, size_t ws_size,
                              hipStream_t stream) {
  const float* x     = (const float*)d_in[0];
  const int*   lens  = (const int*)  d_in[1];
  const float* W_in  = (const float*)d_in[2];
  const float* b_in  = (const float*)d_in[3];
  const float* W_q   = (const float*)d_in[4];
  const float* b_q   = (const float*)d_in[5];
  const float* W_out = (const float*)d_in[6];
  const float* b_out = (const float*)d_in[7];
  const int*   wsz   = (const int*)  d_in[8];

  float* out  = (float*)d_out;
  float* hbuf = (float*)d_ws;                                // 32 MiB
  _Float16* Wt = (_Float16*)((char*)d_ws + (size_t)kB * kT * kD * sizeof(float));

  convert_w<<<dim3((kIN * kD / 4 + 255) / 256), 256, 0, stream>>>(W_in, Wt);

  dim3 gridA(kD / 128, (kB * kT) / 128);                     // (4, 128)
  in_proj_wmma<<<gridA, 256, 0, stream>>>(x, Wt, b_in, hbuf);

  float* outO = out;
  float* outW = out + (size_t)kB * kT * kO;
  atten_scan<<<dim3(kB), 256, 0, stream>>>(hbuf, lens, W_q, b_q, W_out, b_out,
                                           wsz, outO, outW);
}
